// ExpertChoiceMoE_58025008169406
// MI455X (gfx1250) — matrix operand order repaired and checked by reading, not compiled, not run on hardware
//
#include <hip/hip_runtime.h>
#include <stdint.h>

// Problem constants (B=4, L=4096, D=1024, F=4096, E=16, cap=1024)
constexpr int NTOK = 16384;
constexpr int DIM  = 1024;
constexpr int FF   = 4096;
constexpr int NEXP = 16;
constexpr int CAP  = 1024;

typedef __attribute__((ext_vector_type(16))) __bf16 v16bf;
typedef __attribute__((ext_vector_type(8)))  __bf16 v8bf;
typedef __attribute__((ext_vector_type(8)))  float  v8f;

union Frag { v16bf v; v8bf h[2]; };
struct alignas(8) BF4 { __bf16 a, b, c, d; };

// Native conversion: lets the backend use gfx1250's packed f32->bf16 cvt ops
// instead of a manual shift/round VALU sequence.
__device__ inline __bf16 f2bf(float f) { return (__bf16)f; }

__device__ inline float gelu_exact(float x) {
    return 0.5f * x * (1.0f + erff(x * 0.70710678118654752440f));
}

// CDNA5 async global->LDS copy (ISA 15.18.3 op 98), tracked by ASYNCcnt.
// loff = LDS byte offset (flat-aperture low 32 bits), g = global address.
__device__ inline void async_copy_b128(unsigned loff, const void* g) {
    asm volatile("global_load_async_to_lds_b128 %0, %1, off"
                 :: "v"(loff), "v"(g) : "memory");
}
__device__ inline void wait_async0() {
    asm volatile("s_wait_asynccnt 0" ::: "memory");
}

// ---------------------------------------------------------------------------
// Kernel 1: RMSNorm -> bf16 xn, router logits + softmax probs. 1 block/token.
// ---------------------------------------------------------------------------
__global__ __launch_bounds__(256) void k_rmsnorm_router(
    const float* __restrict__ x, const float* __restrict__ nw,
    const float* __restrict__ rw, const float* __restrict__ rb,
    __bf16* __restrict__ xn_bf, float* __restrict__ logits,
    float* __restrict__ probs)
{
    const int n = blockIdx.x;
    const int t = threadIdx.x;
    __shared__ float red[256];
    __shared__ float xsh[DIM];
    __shared__ float lg[NEXP];

    const float4 v = ((const float4*)(x + (size_t)n * DIM))[t];
    float ss = v.x * v.x + v.y * v.y + v.z * v.z + v.w * v.w;
    red[t] = ss;
    __syncthreads();
    for (int s = 128; s > 0; s >>= 1) {
        if (t < s) red[t] += red[t + s];
        __syncthreads();
    }
    const float rinv = rsqrtf(red[0] * (1.0f / DIM) + 1e-5f);

    const float4 wv = ((const float4*)nw)[t];
    float4 xn;
    xn.x = v.x * rinv * wv.x; xn.y = v.y * rinv * wv.y;
    xn.z = v.z * rinv * wv.z; xn.w = v.w * rinv * wv.w;
    xsh[4 * t + 0] = xn.x; xsh[4 * t + 1] = xn.y;
    xsh[4 * t + 2] = xn.z; xsh[4 * t + 3] = xn.w;
    BF4 pk; pk.a = f2bf(xn.x); pk.b = f2bf(xn.y); pk.c = f2bf(xn.z); pk.d = f2bf(xn.w);
    ((BF4*)(xn_bf + (size_t)n * DIM))[t] = pk;
    __syncthreads();

    // Router partials: thread t -> expert e = t>>4, chunk ch = t&15 (64 dims)
    const int e  = t >> 4;
    const int ch = t & 15;
    const float* w = rw + (size_t)e * DIM + ch * 64;
    const float* xs = xsh + ch * 64;
    float p = 0.f;
#pragma unroll 8
    for (int j = 0; j < 64; ++j) p = fmaf(xs[j], w[j], p);
    red[t] = p;
    __syncthreads();
    if (t < NEXP) {
        float s = 0.f;
#pragma unroll
        for (int c = 0; c < 16; ++c) s += red[t * 16 + c];
        s += rb[t];
        logits[(size_t)n * NEXP + t] = s;
        lg[t] = s;
    }
    __syncthreads();
    if (t < NEXP) {
        float mx = lg[0];
#pragma unroll
        for (int i = 1; i < NEXP; ++i) mx = fmaxf(mx, lg[i]);
        float es = 0.f;
#pragma unroll
        for (int i = 0; i < NEXP; ++i) es += expf(lg[i] - mx);
        probs[(size_t)n * NEXP + t] = expf(lg[t] - mx) / es;
    }
}

// ---------------------------------------------------------------------------
// Kernel 2: per-expert top-CAP radix select + stable compaction.
// grid = NEXP blocks x 1024 threads.
// ---------------------------------------------------------------------------
__global__ __launch_bounds__(1024) void k_select(
    const float* __restrict__ logits, unsigned* __restrict__ keys,
    int* __restrict__ idx, int* __restrict__ pos)
{
    const int e = blockIdx.x;
    const int t = threadIdx.x;
    unsigned* ke = keys + (size_t)e * NTOK;
    int* pe = pos + (size_t)e * NTOK;

    // Build order-preserving sortable keys (descending by float), init pos.
#pragma unroll
    for (int j = 0; j < 16; ++j) {
        const int n = t + j * 1024;
        unsigned u = __float_as_uint(logits[(size_t)n * NEXP + e]);
        u = (u & 0x80000000u) ? ~u : (u | 0x80000000u);
        ke[n] = u;
        pe[n] = -1;
    }
    __syncthreads();

    __shared__ unsigned scnt[1024];
    unsigned prefix = 0, k = CAP;
    for (int bit = 31; bit >= 0; --bit) {
        const unsigned cb = 1u << bit;
        const unsigned hm = ~(cb | (cb - 1u));   // bits strictly above `bit`
        const unsigned want = prefix | cb;
        unsigned cnt = 0;
#pragma unroll
        for (int j = 0; j < 16; ++j) {
            const unsigned key = ke[t + j * 1024];
            cnt += ((key & (hm | cb)) == want) ? 1u : 0u;
        }
        scnt[t] = cnt;
        __syncthreads();
        for (int s = 512; s > 0; s >>= 1) {
            if (t < s) scnt[t] += scnt[t + s];
            __syncthreads();
        }
        const unsigned c1 = scnt[0];
        __syncthreads();
        if (k <= c1) prefix |= cb; else k -= c1;
    }
    const unsigned T = prefix;     // key value of CAP-th largest
    // k = number of ==T elements to keep (lowest token indices first)

    const int base = t * 16;       // contiguous range for stability
    unsigned cg = 0, ceq = 0;
#pragma unroll
    for (int j = 0; j < 16; ++j) {
        const unsigned key = ke[base + j];
        cg  += (key > T)  ? 1u : 0u;
        ceq += (key == T) ? 1u : 0u;
    }

    // exclusive scan of cg
    scnt[t] = cg;
    __syncthreads();
    for (int s = 1; s < 1024; s <<= 1) {
        const unsigned v = (t >= s) ? scnt[t - s] : 0u;
        __syncthreads();
        scnt[t] += v;
        __syncthreads();
    }
    const unsigned exG  = scnt[t] - cg;
    const unsigned Gtot = scnt[1023];
    __syncthreads();

    unsigned p = exG;
#pragma unroll
    for (int j = 0; j < 16; ++j) {
        const int n = base + j;
        if (ke[n] > T) { idx[e * CAP + p] = n; pe[n] = (int)p; ++p; }
    }

    // exclusive scan of ceq
    scnt[t] = ceq;
    __syncthreads();
    for (int s = 1; s < 1024; s <<= 1) {
        const unsigned v = (t >= s) ? scnt[t - s] : 0u;
        __syncthreads();
        scnt[t] += v;
        __syncthreads();
    }
    const unsigned exE = scnt[t] - ceq;
    unsigned q = Gtot + exE;
#pragma unroll
    for (int j = 0; j < 16; ++j) {
        const int n = base + j;
        if (ke[n] == T) {
            if (q < CAP) { idx[e * CAP + q] = n; pe[n] = (int)q; }
            ++q;
        }
    }
}

// ---------------------------------------------------------------------------
// Kernels 3/4: tiled bf16 WMMA GEMM, double-buffered LDS, async A-tile copy.
// MODE 0 (fc1): A = xn gathered via idx, out = gelu(A@W^T + b) -> bf16 h
// MODE 1 (fc2): A = h direct,             out = A@W^T + b      -> fp32 y
// Block: 256 threads = 8 waves (2x4), block tile 128x128, K-step 32.
// Each wave: 64x32 = 4x2 tiles of 16x16 -> 8 v_wmma_f32_16x16x32_bf16/step.
// ---------------------------------------------------------------------------
template <int MODE>
__global__ __launch_bounds__(256) void k_gemm(
    const __bf16* __restrict__ A, const float* __restrict__ W,
    const float* __restrict__ bias, const int* __restrict__ idx,
    __bf16* __restrict__ outb, float* __restrict__ outf,
    int K, int Nn)
{
    const int e  = blockIdx.z;
    const int bm = blockIdx.y;
    const int bn = blockIdx.x;
    const int t  = threadIdx.x;
    const int wid  = t >> 5;
    const int lane = t & 31;
    const int wm = wid >> 2;          // 0..1
    const int wn = wid & 3;           // 0..3
    const int laneM  = lane & 15;
    const int laneHi = lane >> 4;

    __shared__ __bf16 As[2][128][40]; // 40 = 32 + 8 pad (keeps 16B alignment)
    __shared__ __bf16 Bs[2][128][40];

    // Hoist A-row resolution (independent of k0): 2 chunks/thread/stage.
    size_t arow[2];
#pragma unroll
    for (int i = 0; i < 2; ++i) {
        const int linear = t + i * 256;
        const int r = linear >> 2;
        const int grow = bm * 128 + r;
        if constexpr (MODE == 0) {
            arow[i] = (size_t)idx[e * CAP + grow];
        } else {
            arow[i] = (size_t)e * CAP + grow;
        }
    }

    const float* wbase[4];
#pragma unroll
    for (int i = 0; i < 4; ++i) {
        const int linear = t + i * 256;
        const int r  = linear >> 3;
        const int c4 = linear & 7;
        wbase[i] = W + ((size_t)e * Nn + bn * 128 + r) * (size_t)K + c4 * 4;
    }

    v8f acc[4][2] = {};
    float4 bw[4];

    // Issue A tile for stage (k0, buf) via async global->LDS (bf16, no convert).
    auto issueA = [&](int buf, int k0) {
#pragma unroll
        for (int i = 0; i < 2; ++i) {
            const int linear = t + i * 256;
            const int r  = linear >> 2;
            const int c8 = linear & 3;
            const __bf16* g = A + arow[i] * (size_t)K + k0 + c8 * 8;
            const unsigned loff = (unsigned)(uintptr_t)&As[buf][r][c8 * 8];
            async_copy_b128(loff, g);
        }
    };
    // Issue B tile global loads into registers (no wait yet).
    auto issueB = [&](int k0) {
#pragma unroll
        for (int i = 0; i < 4; ++i) {
            bw[i] = *(const float4*)(wbase[i] + k0);
            if (k0 + 64 < K) __builtin_prefetch(wbase[i] + k0 + 64, 0, 0);
        }
    };
    // Convert + commit B tile to LDS (placed after compute to hide loadcnt).
    auto commitB = [&](int buf) {
#pragma unroll
        for (int i = 0; i < 4; ++i) {
            const int linear = t + i * 256;
            const int r  = linear >> 3;
            const int c4 = linear & 7;
            BF4 pk;
            pk.a = f2bf(bw[i].x); pk.b = f2bf(bw[i].y);
            pk.c = f2bf(bw[i].z); pk.d = f2bf(bw[i].w);
            *(BF4*)&Bs[buf][r][c4 * 4] = pk;
        }
    };

    // Prologue: stage 0.
    issueA(0, 0);
    issueB(0);
    commitB(0);
    wait_async0();
    __syncthreads();

    const int S = K / 32;
    for (int s = 0; s < S; ++s) {
        const int cur = s & 1;
        const bool more = (s + 1 < S);
        if (more) {
            issueA(cur ^ 1, (s + 1) * 32);
            issueB((s + 1) * 32);
        }

        // ---- compute stage s from buffer `cur` ----
        Frag a[4];
#pragma unroll
        for (int m = 0; m < 4; ++m) {
            const int row = wm * 64 + m * 16 + laneM;
            const int kb = laneHi * 8;
            a[m].h[0] = *(const v8bf*)&As[cur][row][kb];        // K = kb..kb+7
            a[m].h[1] = *(const v8bf*)&As[cur][row][16 + kb];   // K = 16+kb..
        }
        Frag b[2];
#pragma unroll
        for (int n = 0; n < 2; ++n) {
            const int col = wn * 32 + n * 16 + laneM;
            const int kb = laneHi * 8;
            b[n].h[0] = *(const v8bf*)&Bs[cur][col][kb];
            b[n].h[1] = *(const v8bf*)&Bs[cur][col][16 + kb];
        }
#pragma unroll
        for (int m = 0; m < 4; ++m)
#pragma unroll
            for (int n = 0; n < 2; ++n)
                acc[m][n] = __builtin_amdgcn_wmma_f32_16x16x32_bf16(
                    false, a[m].v, false, b[n].v,
                    (short)0, acc[m][n], false, false);

        if (more) commitB(cur ^ 1);
        wait_async0();
        __syncthreads();
    }

    // ---- epilogue: D layout -> M = i + 8*laneHi, N = laneM ----
#pragma unroll
    for (int m = 0; m < 4; ++m) {
#pragma unroll
        for (int n = 0; n < 2; ++n) {
            const int col = bn * 128 + wn * 32 + n * 16 + laneM;
            const float bv = bias[(size_t)e * Nn + col];
#pragma unroll
            for (int i = 0; i < 8; ++i) {
                const int row = bm * 128 + wm * 64 + m * 16 + laneHi * 8 + i;
                float v = acc[m][n][i] + bv;
                if constexpr (MODE == 0) {
                    v = gelu_exact(v);
                    outb[((size_t)e * CAP + row) * (size_t)Nn + col] = f2bf(v);
                } else {
                    outf[((size_t)e * CAP + row) * (size_t)Nn + col] = v;
                }
            }
        }
    }
}

// ---------------------------------------------------------------------------
// Kernel 5: deterministic gated combine + normalize. 1 block/token.
// ---------------------------------------------------------------------------
__global__ __launch_bounds__(256) void k_combine(
    const float* __restrict__ y, const float* __restrict__ probs,
    const int* __restrict__ pos, float* __restrict__ out)
{
    const int n = blockIdx.x;
    const int t = threadIdx.x;            // covers d = 4t..4t+3
    float4 acc = {0.f, 0.f, 0.f, 0.f};
    float gs = 0.f;
#pragma unroll
    for (int e = 0; e < NEXP; ++e) {
        const int p = pos[(size_t)e * NTOK + n];
        if (p >= 0) {
            const float g = probs[(size_t)n * NEXP + e];
            gs += g;
            const float4 yv = *(const float4*)(
                y + (((size_t)e * CAP + p) * DIM) + t * 4);
            acc.x = fmaf(g, yv.x, acc.x);
            acc.y = fmaf(g, yv.y, acc.y);
            acc.z = fmaf(g, yv.z, acc.z);
            acc.w = fmaf(g, yv.w, acc.w);
        }
    }
    const float inv = (gs > 0.f) ? 1.f / gs : 0.f;
    float4 o; o.x = acc.x * inv; o.y = acc.y * inv; o.z = acc.z * inv; o.w = acc.w * inv;
    ((float4*)(out + (size_t)n * DIM))[t] = o;
}

// ---------------------------------------------------------------------------
extern "C" void kernel_launch(void* const* d_in, const int* in_sizes, int n_in,
                              void* d_out, int out_size, void* d_ws, size_t ws_size,
                              hipStream_t stream)
{
    const float* x  = (const float*)d_in[0];
    const float* nw = (const float*)d_in[1];
    const float* rw = (const float*)d_in[2];
    const float* rb = (const float*)d_in[3];
    const float* w1 = (const float*)d_in[4];
    const float* b1 = (const float*)d_in[5];
    const float* w2 = (const float*)d_in[6];
    const float* b2 = (const float*)d_in[7];
    float* out = (float*)d_out;

    size_t off = 0;
    auto alloc = [&](size_t bytes) -> void* {
        void* p = (char*)d_ws + off;
        off += (bytes + 255) & ~(size_t)255;
        return p;
    };
    __bf16*   xn_bf  = (__bf16*)  alloc((size_t)NTOK * DIM * 2);
    float*    logits = (float*)   alloc((size_t)NTOK * NEXP * 4);
    float*    probs  = (float*)   alloc((size_t)NTOK * NEXP * 4);
    unsigned* keys   = (unsigned*)alloc((size_t)NEXP * NTOK * 4);
    int*      idx    = (int*)     alloc((size_t)NEXP * CAP * 4);
    int*      pos    = (int*)     alloc((size_t)NEXP * NTOK * 4);
    __bf16*   h_bf   = (__bf16*)  alloc((size_t)NEXP * CAP * FF * 2);
    float*    y      = (float*)   alloc((size_t)NEXP * CAP * DIM * 4);
    (void)ws_size; (void)in_sizes; (void)n_in; (void)out_size;

    k_rmsnorm_router<<<NTOK, 256, 0, stream>>>(x, nw, rw, rb, xn_bf, logits, probs);
    k_select<<<NEXP, 1024, 0, stream>>>(logits, keys, idx, pos);
    // fc1: M=CAP, N=FF, K=DIM
    k_gemm<0><<<dim3(FF / 128, CAP / 128, NEXP), 256, 0, stream>>>(
        xn_bf, w1, b1, idx, h_bf, nullptr, DIM, FF);
    // fc2: M=CAP, N=DIM, K=FF
    k_gemm<1><<<dim3(DIM / 128, CAP / 128, NEXP), 256, 0, stream>>>(
        h_bf, w2, b2, nullptr, nullptr, y, FF, DIM);
    k_combine<<<NTOK, 256, 0, stream>>>(y, probs, pos, out);
}
